// GraphEdgeVertPredHighway_78408922956222
// MI455X (gfx1250) — matrix operand order repaired and checked by reading, not compiled, not run on hardware
//
#include <hip/hip_runtime.h>
#include <hip/hip_bf16.h>

// GraphEdgeVertPredHighway on gfx1250 (MI455X).
// B=16, N=64, FV=64, FE=4, DV=DE=256, NL=6.
//
// L2-bound problem. All GEMMs via v_wmma_f32_16x16x32_bf16 (f32 accumulate).
// Weight-chunk staging uses CDNA5 async global->LDS copies
// (GLOBAL_LOAD_ASYNC_TO_LDS_B128 / ASYNCcnt); builtin signature per hipcc
// diagnostic: (int4 __device__* src, int4 __shared__* dst, int, int).

typedef __attribute__((ext_vector_type(16))) __bf16 v16bf;
typedef __attribute__((ext_vector_type(8)))  __bf16 v8bf;
typedef __attribute__((ext_vector_type(8)))  float  v8f;
typedef __attribute__((ext_vector_type(4)))  int    v4i;

#define NB   16
#define NN   64
#define DE_  256
#define DV_  256
#define KPAD 40   // 80B row stride = 20 dwords -> 16 lanes hit 16 distinct banks

#if defined(__has_builtin)
#if __has_builtin(__builtin_amdgcn_global_load_async_to_lds_b128)
#define ASYNC_STAGE 1
#endif
#endif
#ifndef ASYNC_STAGE
#define ASYNC_STAGE 0
#endif

#if ASYNC_STAGE
typedef __attribute__((address_space(1))) v4i gv4i;   // global int4
typedef __attribute__((address_space(3))) v4i lv4i;   // LDS int4
#endif

// ---------------------------------------------------------------------------
static __device__ __forceinline__ unsigned int pack2bf(float x, float y) {
  union { __bf16 h; unsigned short u; } a, b;
  a.h = (__bf16)x; b.h = (__bf16)y;
  return (unsigned int)a.u | ((unsigned int)b.u << 16);
}

static __device__ __forceinline__ v16bf ld_frag(const __bf16* p0,
                                                const __bf16* p1) {
  v8bf lo = *(const v8bf*)p0;   // 16B aligned -> ds_load_b128
  v8bf hi = *(const v8bf*)p1;
  return __builtin_shufflevector(lo, hi, 0, 1, 2, 3, 4, 5, 6, 7,
                                 8, 9, 10, 11, 12, 13, 14, 15);
}

// Wait for this wave's outstanding async LDS copies (ASYNCcnt); the caller's
// __syncthreads then globalizes completion across the workgroup.
static __device__ __forceinline__ void wait_async() {
#if ASYNC_STAGE
#if defined(__has_builtin) && __has_builtin(__builtin_amdgcn_s_wait_asynccnt)
  __builtin_amdgcn_s_wait_asynccnt(0);
#else
  asm volatile("s_wait_asynccnt 0x0" ::: "memory");
#endif
#endif
}

// ---------------------------------------------------------------------------
// Per-chunk WMMA step. Alds: 64-row sub-block, row-major [.][KPAD];
// Bldsw: column-major [256][KPAD]. Wave w: m-tile = w&3, n-half = w>>2.
// Fragment maps per CDNA5 ISA 7.12.2 (wave32):
//   A 16x32 bf16 : e<8 -> K=half*8+e ; e>=8 -> K=16+half*8+(e-8)
//   B 32x16 bf16 : same K map as A ; col = lane&15
//   C/D 16x16 f32: m = half*8+r ; n = lane&15
// ---------------------------------------------------------------------------
__device__ __forceinline__ void wmma_tiles(const __bf16 (*Alds)[KPAD],
                                           const __bf16 (&Bldsw)[256][KPAD],
                                           int lane, int mtile, int nbase,
                                           v8f acc[8]) {
  const int half = lane >> 4;
  const int mrow = mtile * 16 + (lane & 15);
  const v16bf a = ld_frag(&Alds[mrow][half * 8], &Alds[mrow][16 + half * 8]);

#pragma unroll
  for (int t = 0; t < 8; ++t) {
    const int col = nbase + t * 16 + (lane & 15);
    const v16bf bb = ld_frag(&Bldsw[col][half * 8], &Bldsw[col][16 + half * 8]);
    acc[t] = __builtin_amdgcn_wmma_f32_16x16x32_bf16(
        false, a, false, bb, (short)0, acc[t], false, false);
  }
}

__device__ __forceinline__ void zero_acc(v8f acc[8]) {
#pragma unroll
  for (int x = 0; x < 8; ++x)
#pragma unroll
    for (int r = 0; r < 8; ++r) acc[x][r] = 0.0f;
}

// Stage one pre-swizzled bf16 weight chunk (layout [col 256][k 32]) into LDS.
// Thread t copies column t: 64B. Async DMA path when available.
__device__ __forceinline__ void stage_Bbf(__bf16 (&Bldsw)[256][KPAD],
                                          const __bf16* __restrict__ Wbf,
                                          int kc, int t) {
#if ASYNC_STAGE
  const __bf16* srcp = Wbf + (size_t)kc * 8192 + t * 32;
  __bf16* dstp = &Bldsw[t][0];
#pragma unroll
  for (int s = 0; s < 4; ++s)
    __builtin_amdgcn_global_load_async_to_lds_b128((gv4i*)srcp + s,
                                                   (lv4i*)dstp + s, 0, 0);
#else
  const uint4* src = (const uint4*)(Wbf + (size_t)kc * 8192 + t * 32);
#pragma unroll
  for (int s = 0; s < 4; ++s)
    *(uint4*)&Bldsw[t][s * 8] = src[s];
#endif
}

// ---------------------------------------------------------------------------
// Weight convert + swizzle: W f32 [K][256] -> Wbf chunk-major:
//   Wbf[kc*8192 + d*32 + k] = bf16(W[(kc*32+k)*256 + d])
// Grid: K/32 blocks, 256 threads (thread = column d).
// ---------------------------------------------------------------------------
__global__ void conv_weight_kernel(const float* __restrict__ W,
                                   __bf16* __restrict__ Wbf) {
  const int kc = blockIdx.x;
  const int d = threadIdx.x;
  __bf16* out = Wbf + (size_t)kc * 8192 + d * 32;
#pragma unroll
  for (int kp = 0; kp < 16; ++kp) {
    const float x0 = W[(kc * 32 + kp * 2 + 0) * 256 + d];
    const float x1 = W[(kc * 32 + kp * 2 + 1) * 256 + d];
    *(unsigned int*)&out[kp * 2] = pack2bf(x0, x1);
  }
}

// ---------------------------------------------------------------------------
// e_v = A[b] (64 x K, row-major f32) @ Wbf (K x 256) + bias -> Out[b]
// ---------------------------------------------------------------------------
__global__ void ev_gemm_kernel(const float* __restrict__ A, int K,
                               const __bf16* __restrict__ Wbf,
                               const float* __restrict__ bias,
                               float* __restrict__ Out) {
  const int b = blockIdx.x;
  const int t = threadIdx.x;
  const int lane = t & 31, wave = t >> 5;
  const int mtile = wave & 3, nbase = (wave >> 2) * 128;

  __shared__ alignas(16) __bf16 Alds[64][KPAD];
  __shared__ alignas(16) __bf16 Bldsw[256][KPAD];

  v8f acc[8];
  zero_acc(acc);

  const float* Ab = A + (size_t)b * 64 * K;
  const int nchunks = K >> 5;
  for (int kc = 0; kc < nchunks; ++kc) {
    stage_Bbf(Bldsw, Wbf, kc, t);
#pragma unroll
    for (int s = 0; s < 4; ++s) {           // 1024 bf16 pairs of A
      const int p = t + s * 256;
      const int j = p >> 4, kp = p & 15;
      const float2 x = *(const float2*)&Ab[j * K + kc * 32 + kp * 2];
      *(unsigned int*)&Alds[j][kp * 2] = pack2bf(x.x, x.y);
    }
    if (kc + 1 < nchunks) {                 // global_prefetch next chunk
      __builtin_prefetch(Wbf + (size_t)(kc + 1) * 8192 + t * 32, 0, 3);
      if (t < 64) __builtin_prefetch(&Ab[t * K + (kc + 1) * 32], 0, 3);
    }
    wait_async();
    __syncthreads();
    wmma_tiles(&Alds[0], Bldsw, lane, mtile, nbase, acc);
    __syncthreads();
  }

  const int half = lane >> 4, nloc = lane & 15;
#pragma unroll
  for (int tt = 0; tt < 8; ++tt) {
    const int n = nbase + tt * 16 + nloc;
#pragma unroll
    for (int r = 0; r < 8; ++r) {
      const int m = mtile * 16 + half * 8 + r;
      Out[(size_t)(b * 64 + m) * 256 + n] = acc[tt][r] + bias[n];
    }
  }
}

// ---------------------------------------------------------------------------
// Inner edge layer, in place on e_cur. One block per (b, i-pair): processes
// i0 = 2*blockIdx.y and i1 = i0+1 with dual accumulators so each staged
// weight chunk serves two 64x256 tiles. In-place safe: the block reads only
// the two tiles it later writes; barriers separate phases; each output
// element is owned by exactly one thread.
// ---------------------------------------------------------------------------
__global__ void edge_layer_kernel(float* __restrict__ e_cur,
                                  const __bf16* __restrict__ w_e_bf,
                                  const float* __restrict__ b_e,
                                  const float* __restrict__ e_v,
                                  int* __restrict__ per_v) {
  const int b = blockIdx.x, i0 = blockIdx.y * 2, i1 = i0 + 1;
  const int t = threadIdx.x;
  const int lane = t & 31, wave = t >> 5;
  const int mtile = wave & 3, nbase = (wave >> 2) * 128;

  __shared__ alignas(16) __bf16 Alds[128][KPAD];   // rows 0-63: i0, 64-127: i1
  __shared__ alignas(16) __bf16 Bldsw[256][KPAD];

  v8f acc0[8], acc1[8];
  zero_acc(acc0);
  zero_acc(acc1);

  float* A0 = e_cur + ((size_t)(b * 64 + i0) * 64) * 256;
  float* A1 = e_cur + ((size_t)(b * 64 + i1) * 64) * 256;
  for (int kc = 0; kc < 8; ++kc) {          // K = 256
    stage_Bbf(Bldsw, w_e_bf, kc, t);
#pragma unroll
    for (int s = 0; s < 4; ++s) {
      const int p = t + s * 256;
      const int j = p >> 4, kp = p & 15;
      const float2 x0 = *(const float2*)&A0[j * 256 + kc * 32 + kp * 2];
      const float2 x1 = *(const float2*)&A1[j * 256 + kc * 32 + kp * 2];
      *(unsigned int*)&Alds[j][kp * 2] = pack2bf(x0.x, x0.y);
      *(unsigned int*)&Alds[64 + j][kp * 2] = pack2bf(x1.x, x1.y);
    }
    if (kc + 1 < 8) {
      __builtin_prefetch(w_e_bf + (size_t)(kc + 1) * 8192 + t * 32, 0, 3);
      if (t < 64) {
        __builtin_prefetch(&A0[t * 256 + (kc + 1) * 32], 0, 3);
        __builtin_prefetch(&A1[t * 256 + (kc + 1) * 32], 0, 3);
      }
    }
    wait_async();
    __syncthreads();
    wmma_tiles(&Alds[0], Bldsw, lane, mtile, nbase, acc0);
    wmma_tiles(&Alds[64], Bldsw, lane, mtile, nbase, acc1);
    __syncthreads();
  }

  const int half = lane >> 4, nloc = lane & 15;
#pragma unroll
  for (int ii = 0; ii < 2; ++ii) {
    const int i = ii ? i1 : i0;
    const v8f* acc = ii ? acc1 : acc0;
    const float* evI = e_v + (size_t)(b * 64 + i) * 256;
#pragma unroll
    for (int tt = 0; tt < 8; ++tt) {
      const int n = nbase + tt * 16 + nloc;
      const float cbias = b_e[n] + evI[n];
#pragma unroll
      for (int r = 0; r < 8; ++r) {
        const int m = mtile * 16 + half * 8 + r;   // j index
        float val = acc[tt][r] + cbias + e_v[(size_t)(b * 64 + m) * 256 + n];
        float rl = val > 0.0f ? val : 0.0f;
        atomicMax(per_v + (size_t)(b * 64 + m) * 256 + n, __float_as_int(rl));
        const size_t g = ((size_t)(b * 64 + i) * 64 + m) * 256 + n;
        e_cur[g] = rl + e_cur[g];                  // residual, in place
      }
    }
  }
}

// ---------------------------------------------------------------------------
// First edge layer: K = FE = 4 -> VALU. e_t[b,i,j,f] = e_raw[b,f,j,i].
// ---------------------------------------------------------------------------
__global__ void edge_first_kernel(const float* __restrict__ e_raw,
                                  const float* __restrict__ w_e1,
                                  const float* __restrict__ b_e1,
                                  const float* __restrict__ e_v,
                                  float* __restrict__ e_cur,
                                  int* __restrict__ per_v) {
  const int b = blockIdx.x, i = blockIdx.y;
  const int t = threadIdx.x;       // d
  __shared__ float Ecol[64][4];
  __shared__ float We1s[4][256];
  __shared__ float EvI[256];

  {
    const int j = t >> 2, f = t & 3;
    Ecol[j][f] = e_raw[((size_t)(b * 4 + f) * 64 + j) * 64 + i];
  }
#pragma unroll
  for (int f = 0; f < 4; ++f) We1s[f][t] = w_e1[f * 256 + t];
  EvI[t] = e_v[(size_t)(b * 64 + i) * 256 + t];
  __syncthreads();

  const float base = b_e1[t] + EvI[t];
  for (int j = 0; j < 64; ++j) {
    float val = base + e_v[(size_t)(b * 64 + j) * 256 + t];
#pragma unroll
    for (int f = 0; f < 4; ++f) val += Ecol[j][f] * We1s[f][t];
    float rl = val > 0.0f ? val : 0.0f;
    atomicMax(per_v + (size_t)(b * 64 + j) * 256 + t, __float_as_int(rl));
    e_cur[((size_t)(b * 64 + i) * 64 + j) * 256 + t] = rl;
  }
}

// ---------------------------------------------------------------------------
// Vertex update: A = concat([per_v (256), vsrc (Kv)]) @ w_v + b_v
//   first=1: v_cur = relu(.), highway = v_cur
//   first=0: v_cur = relu(.) + v_cur, highway += v_cur
// Chunk and concat boundaries are 32-aligned -> pairs never straddle.
// ---------------------------------------------------------------------------
__global__ void vertex_kernel(const float* __restrict__ per_v,
                              const float* __restrict__ vsrc, int Kv,
                              const __bf16* __restrict__ w_v_bf,
                              const float* __restrict__ b_v,
                              float* __restrict__ v_cur,
                              float* __restrict__ highway, int first) {
  const int b = blockIdx.x;
  const int t = threadIdx.x;
  const int lane = t & 31, wave = t >> 5;
  const int mtile = wave & 3, nbase = (wave >> 2) * 128;

  __shared__ alignas(16) __bf16 Alds[64][KPAD];
  __shared__ alignas(16) __bf16 Bldsw[256][KPAD];

  v8f acc[8];
  zero_acc(acc);

  const int Ktot = 256 + Kv;
  const int nchunks = Ktot >> 5;
  for (int kc = 0; kc < nchunks; ++kc) {
    stage_Bbf(Bldsw, w_v_bf, kc, t);
#pragma unroll
    for (int s = 0; s < 4; ++s) {
      const int p = t + s * 256;
      const int j = p >> 4, kp = p & 15;
      const int kg = kc * 32 + kp * 2;
      float2 x;
      if (kg < 256) x = *(const float2*)&per_v[(size_t)(b * 64 + j) * 256 + kg];
      else          x = *(const float2*)&vsrc[(size_t)(b * 64 + j) * Kv + (kg - 256)];
      *(unsigned int*)&Alds[j][kp * 2] = pack2bf(x.x, x.y);
    }
    if (kc + 1 < nchunks)
      __builtin_prefetch(w_v_bf + (size_t)(kc + 1) * 8192 + t * 32, 0, 3);
    wait_async();
    __syncthreads();
    wmma_tiles(&Alds[0], Bldsw, lane, mtile, nbase, acc);
    __syncthreads();
  }

  const int half = lane >> 4, nloc = lane & 15;
#pragma unroll
  for (int tt = 0; tt < 8; ++tt) {
    const int n = nbase + tt * 16 + nloc;
#pragma unroll
    for (int r = 0; r < 8; ++r) {
      const int m = mtile * 16 + half * 8 + r;
      const size_t idx = (size_t)(b * 64 + m) * 256 + n;
      float val = acc[tt][r] + b_v[n];
      float rl = val > 0.0f ? val : 0.0f;
      float vc = first ? rl : (rl + v_cur[idx]);
      v_cur[idx] = vc;
      highway[idx] = first ? vc : (highway[idx] + vc);
    }
  }
}

// ---------------------------------------------------------------------------
// Head: out[idx] = relu(highway[idx,:]) . out_w + out_b   (idx over B*N)
// ---------------------------------------------------------------------------
__global__ void head_kernel(const float* __restrict__ highway,
                            const float* __restrict__ out_w,
                            const float* __restrict__ out_b,
                            float* __restrict__ out) {
  const int idx = blockIdx.x;
  const int t = threadIdx.x;
  __shared__ float red[256];
  float h = highway[(size_t)idx * 256 + t];
  red[t] = (h > 0.0f ? h : 0.0f) * out_w[t];
  __syncthreads();
  for (int s = 128; s > 0; s >>= 1) {
    if (t < s) red[t] += red[t + s];
    __syncthreads();
  }
  if (t == 0) out[idx] = red[0] + out_b[0];
}

// ---------------------------------------------------------------------------
extern "C" void kernel_launch(void* const* d_in, const int* in_sizes, int n_in,
                              void* d_out, int out_size, void* d_ws,
                              size_t ws_size, hipStream_t stream) {
  (void)in_sizes; (void)n_in; (void)out_size; (void)ws_size;

  const float* e     = (const float*)d_in[0];   // [16,4,64,64]
  const float* v     = (const float*)d_in[1];   // [16,64,64]
  const float* w_ev1 = (const float*)d_in[2];   // [64,256]
  const float* b_ev1 = (const float*)d_in[3];
  const float* w_e1  = (const float*)d_in[4];   // [4,256]
  const float* b_e1  = (const float*)d_in[5];
  const float* w_v1  = (const float*)d_in[6];   // [320,256]
  const float* b_v1  = (const float*)d_in[7];
  const float* w_ev  = (const float*)d_in[8];   // [6,256,256]
  const float* b_ev  = (const float*)d_in[9];
  const float* w_e   = (const float*)d_in[10];  // [6,256,256]
  const float* b_e   = (const float*)d_in[11];
  const float* w_v   = (const float*)d_in[12];  // [6,512,256]
  const float* b_v   = (const float*)d_in[13];
  const float* out_w = (const float*)d_in[14];  // [256,1]
  const float* out_b = (const float*)d_in[15];

  float* ws      = (float*)d_ws;
  float* e_cur   = ws;                                    // 16*64*64*256
  float* e_v     = e_cur + (size_t)NB * NN * NN * DE_;    // 16*64*256
  float* per_v   = e_v   + (size_t)NB * NN * DE_;
  float* v_cur   = per_v + (size_t)NB * NN * DE_;
  float* highway = v_cur + (size_t)NB * NN * DV_;

  // bf16 weight arena (16B-aligned: float region ends on 16B boundary)
  __bf16* warena   = (__bf16*)(highway + (size_t)NB * NN * DV_);
  __bf16* w_ev1_bf = warena;                 // 64*256
  __bf16* w_v1_bf  = w_ev1_bf + 16384;       // 320*256
  __bf16* w_ev_bf  = w_v1_bf + 81920;        // 6*256*256
  __bf16* w_e_bf   = w_ev_bf + 393216;       // 6*256*256
  __bf16* w_v_bf   = w_e_bf + 393216;        // 6*512*256

  const dim3 blk(256);
  const dim3 grid_b(NB);
  const dim3 grid_bn(NB, NN);
  const dim3 grid_bn2(NB, NN / 2);
  const size_t pv_bytes = (size_t)NB * NN * DE_ * sizeof(float);

  // ---- weight convert + swizzle (once per call) ----
  conv_weight_kernel<<<dim3(2), blk, 0, stream>>>(w_ev1, w_ev1_bf);
  conv_weight_kernel<<<dim3(10), blk, 0, stream>>>(w_v1, w_v1_bf);
  for (int l = 0; l < 6; ++l) {
    conv_weight_kernel<<<dim3(8), blk, 0, stream>>>(w_ev + l * 65536,
                                                    w_ev_bf + l * 65536);
    conv_weight_kernel<<<dim3(8), blk, 0, stream>>>(w_e + l * 65536,
                                                    w_e_bf + l * 65536);
    conv_weight_kernel<<<dim3(16), blk, 0, stream>>>(w_v + l * 131072,
                                                     w_v_bf + l * 131072);
  }

  // ---- layer 1 ----
  ev_gemm_kernel<<<grid_b, blk, 0, stream>>>(v, 64, w_ev1_bf, b_ev1, e_v);
  (void)hipMemsetAsync(per_v, 0, pv_bytes, stream);
  edge_first_kernel<<<grid_bn, blk, 0, stream>>>(e, w_e1, b_e1, e_v, e_cur,
                                                 (int*)per_v);
  vertex_kernel<<<grid_b, blk, 0, stream>>>(per_v, v, 64, w_v1_bf, b_v1,
                                            v_cur, highway, 1);

  // ---- inner layers ----
  for (int l = 0; l < 6; ++l) {
    ev_gemm_kernel<<<grid_b, blk, 0, stream>>>(v_cur, 256, w_ev_bf + l * 65536,
                                               b_ev + l * 256, e_v);
    (void)hipMemsetAsync(per_v, 0, pv_bytes, stream);
    edge_layer_kernel<<<grid_bn2, blk, 0, stream>>>(e_cur, w_e_bf + l * 65536,
                                                    b_e + l * 256, e_v,
                                                    (int*)per_v);
    vertex_kernel<<<grid_b, blk, 0, stream>>>(per_v, v_cur, 256,
                                              w_v_bf + l * 131072,
                                              b_v + l * 256,
                                              v_cur, highway, 0);
  }

  // ---- head ----
  head_kernel<<<dim3(NB * NN), blk, 0, stream>>>(highway, out_w, out_b,
                                                 (float*)d_out);
}
